// OptimizedGATModel_54769422959047
// MI455X (gfx1250) — matrix operand order repaired and checked by reading, not compiled, not run on hardware
//
#include <hip/hip_runtime.h>
#include <hip/hip_bf16.h>

// ---------------------------------------------------------------------------
// CDNA5 (gfx1250) fused GAT mixture model.
// Big GEMMs use v_wmma_f32_16x16x32_bf16 (wave32, 16x16 tiles, K=32/instr).
// ---------------------------------------------------------------------------

typedef __attribute__((ext_vector_type(16))) __bf16 v16bf;
typedef __attribute__((ext_vector_type(8)))  float  v8f;

union Frag16 { unsigned int u[8]; v16bf v; };
union AccF   { float f[8];        v8f   v; };

#define BATCH   16384
#define NNODE   7
#define NBROWS  (BATCH * NNODE)   // 114688
#define HIDC    256
#define GDIM    1920
#define HOUT    80                 // 32 (base) + 32 (reg) + 16 (res)
#define NEGV    (-1.0e9f)

__device__ __forceinline__ unsigned short f2bf(float f) {
    unsigned int x = __float_as_uint(f);
    unsigned int r = (x + 0x7fffu + ((x >> 16) & 1u)) >> 16;  // RNE
    return (unsigned short)r;
}

__device__ __forceinline__ float wred(float v) {
    #pragma unroll
    for (int o = 16; o > 0; o >>= 1) v += __shfl_xor(v, o, 32);
    return v;
}

// ---------------------------------------------------------------------------
// Prep: bf16-transpose weights into ws; edge mask.
// ---------------------------------------------------------------------------
__global__ void k_prep(const float* __restrict__ W1, const float* __restrict__ SKw,
                       const float* __restrict__ W2, const float* __restrict__ bw1,
                       const float* __restrict__ rw1, const float* __restrict__ rsw1,
                       const float* __restrict__ elog,
                       unsigned short* __restrict__ W1t, unsigned short* __restrict__ SKt,
                       unsigned short* __restrict__ W2t, unsigned short* __restrict__ Wht,
                       float* __restrict__ maskws, float* __restrict__ maskout) {
    int stride = gridDim.x * blockDim.x;
    int tid0 = blockIdx.x * blockDim.x + threadIdx.x;
    for (int i = tid0; i < HIDC * HIDC; i += stride) {
        int n = i >> 8, k = i & 255;
        W1t[i] = f2bf(W1[k * HIDC + n]);
        SKt[i] = f2bf(SKw[k * HIDC + n]);
        W2t[i] = f2bf(W2[k * HIDC + n]);
    }
    for (int i = tid0; i < HOUT * GDIM; i += stride) {
        int n = i / GDIM, k = i % GDIM;
        float v = (n < 32) ? bw1[k * 32 + n]
                : (n < 64) ? rw1[k * 32 + (n - 32)]
                           : rsw1[k * 16 + (n - 64)];
        Wht[i] = f2bf(v);
    }
    for (int i = tid0; i < NNODE * NNODE; i += stride) {
        int r = i / NNODE, c = i % NNODE;
        float m = (r == c) ? 1.0f : 1.0f / (1.0f + expf(-elog[i]));
        maskws[i] = m;
        maskout[i] = m;
    }
}

// ---------------------------------------------------------------------------
// GEMM: (M x 256) fp32 @ bf16-transposed weights -> fp32.
// dual=1: 8 waves, one 16-row tile, waves 0-3 -> C0 (B0), waves 4-7 -> C1 (B1)
// dual=0: 8 waves, two 16-row tiles of C0 (B0 only), 32 rows per block.
// ---------------------------------------------------------------------------
__global__ __launch_bounds__(256, 2)
void k_gemm256(const float* __restrict__ A, const unsigned short* __restrict__ B0,
               const unsigned short* __restrict__ B1, float* __restrict__ C0,
               float* __restrict__ C1, int dual) {
    __shared__ unsigned short ap[32 * HIDC];   // up to 16 KB bf16 A-panel
    const int tid = threadIdx.x;
    const int bx = blockIdx.x;
    const int rowsPer = dual ? 16 : 32;
    const int rowBase = bx * rowsPer;

    for (int i = tid; i < rowsPer * HIDC; i += 256)
        ap[i] = f2bf(A[(size_t)(rowBase + (i >> 8)) * HIDC + (i & 255)]);
    __syncthreads();

    const int wave = tid >> 5, lane = tid & 31;
    const int half = lane >> 4, l16 = lane & 15;

    const unsigned short* Bt;
    float* Cout;
    int colBase = (wave & 3) * 64, mOff, outRow;
    if (dual) {
        Bt   = (wave < 4) ? B0 : B1;
        Cout = (wave < 4) ? C0 : C1;
        mOff = 0; outRow = rowBase;
    } else {
        Bt = B0; Cout = C0;
        mOff = (wave < 4) ? 0 : 16;
        outRow = rowBase + mOff;
    }

    AccF acc[4];
    #pragma unroll
    for (int s = 0; s < 4; ++s)
        #pragma unroll
        for (int v = 0; v < 8; ++v) acc[s].f[v] = 0.0f;

    for (int kb = 0; kb < 8; ++kb) {
        Frag16 a;
        #pragma unroll
        for (int v = 0; v < 8; ++v) {
            int kp = kb * 32 + 16 * half + 2 * v;
            a.u[v] = *(const unsigned int*)&ap[(mOff + l16) * HIDC + kp];
        }
        #pragma unroll
        for (int s = 0; s < 4; ++s) {
            Frag16 b;
            int n = colBase + s * 16 + l16;
            #pragma unroll
            for (int v = 0; v < 8; ++v) {
                int kp = kb * 32 + 16 * half + 2 * v;
                b.u[v] = *(const unsigned int*)&Bt[(size_t)n * HIDC + kp];
            }
            acc[s].v = __builtin_amdgcn_wmma_f32_16x16x32_bf16(
                false, a.v, false, b.v, (short)0, acc[s].v, false, false);
        }
    }

    #pragma unroll
    for (int s = 0; s < 4; ++s)
        #pragma unroll
        for (int v = 0; v < 8; ++v)
            Cout[(size_t)(outRow + v + 8 * half) * HIDC + colBase + s * 16 + l16] =
                acc[s].f[v];
}

// ---------------------------------------------------------------------------
// Head GEMM: (B x 1920) @ (1920 x 80) via 16x1920 bf16 LDS panel, 5 waves.
// ---------------------------------------------------------------------------
__global__ __launch_bounds__(160, 1)
void k_gemm_head(const float* __restrict__ A, const unsigned short* __restrict__ Bt,
                 float* __restrict__ C) {
    __shared__ unsigned short ap[16 * GDIM];   // 60 KB
    const int tid = threadIdx.x;
    const int bx = blockIdx.x;

    for (int i = tid; i < 16 * GDIM; i += 160) {
        int r = i / GDIM, c = i % GDIM;
        ap[i] = f2bf(A[(size_t)(bx * 16 + r) * GDIM + c]);
    }
    __syncthreads();

    const int wave = tid >> 5, lane = tid & 31;
    const int half = lane >> 4, l16 = lane & 15;
    const int n0 = wave * 16;

    AccF acc;
    #pragma unroll
    for (int v = 0; v < 8; ++v) acc.f[v] = 0.0f;

    for (int kb = 0; kb < GDIM / 32; ++kb) {
        Frag16 a, b;
        #pragma unroll
        for (int v = 0; v < 8; ++v) {
            int kpa = kb * 32 + 16 * half + 2 * v;
            a.u[v] = *(const unsigned int*)&ap[l16 * GDIM + kpa];
            int kpb = kb * 32 + 16 * half + 2 * v;
            b.u[v] = *(const unsigned int*)&Bt[(size_t)(n0 + l16) * GDIM + kpb];
        }
        acc.v = __builtin_amdgcn_wmma_f32_16x16x32_bf16(
            false, a.v, false, b.v, (short)0, acc.v, false, false);
    }
    #pragma unroll
    for (int v = 0; v < 8; ++v)
        C[(size_t)(bx * 16 + v + 8 * half) * HOUT + n0 + l16] = acc.f[v];
}

// ---------------------------------------------------------------------------
// Fused attention + residual + LayerNorm + ReLU. One wave per batch.
// ---------------------------------------------------------------------------
__global__ __launch_bounds__(256, 2)
void k_attn(const float* __restrict__ Hm,       // (B*7,256)  x@W
            const float* __restrict__ Resid,    // skip (L1) or h (L2)
            const float* __restrict__ res_bias, // skip_b or nullptr
            const float* __restrict__ maskp,    // 49 floats
            const float* __restrict__ a_src, const float* __restrict__ a_dst,
            const float* __restrict__ ln_w, const float* __restrict__ ln_b,
            float* __restrict__ OutH, int batchStride,
            const float* __restrict__ context,  // nullptr for L1
            float* __restrict__ attn_out) {     // nullptr for L1
    __shared__ float sc[8][260];   // per-wave: [0..195] alpha, [200..227] s_src, [228..255] s_dst
    const int tid = threadIdx.x, wv = tid >> 5, lane = tid & 31;
    const long long b  = (long long)blockIdx.x * 8 + wv;
    const long long r0 = b * NNODE;

    float hv[7][8];
    #pragma unroll
    for (int n = 0; n < 7; ++n)
        #pragma unroll
        for (int t = 0; t < 8; ++t)
            hv[n][t] = Hm[(r0 + n) * HIDC + lane + 32 * t];

    float asv[8], adv[8];
    #pragma unroll
    for (int t = 0; t < 8; ++t) {
        int c = lane + 32 * t;
        asv[t] = a_src[(t >> 1) * 64 + (c & 63)];
        adv[t] = a_dst[(t >> 1) * 64 + (c & 63)];
    }

    #pragma unroll
    for (int n = 0; n < 7; ++n) {
        float ps[4] = {0, 0, 0, 0}, pd[4] = {0, 0, 0, 0};
        #pragma unroll
        for (int t = 0; t < 8; ++t) {
            ps[t >> 1] += hv[n][t] * asv[t];
            pd[t >> 1] += hv[n][t] * adv[t];
        }
        #pragma unroll
        for (int h = 0; h < 4; ++h) { ps[h] = wred(ps[h]); pd[h] = wred(pd[h]); }
        if (lane == 0) {
            #pragma unroll
            for (int h = 0; h < 4; ++h) {
                sc[wv][200 + n * 4 + h] = ps[h];
                sc[wv][228 + n * 4 + h] = pd[h];
            }
        }
    }
    __syncthreads();

    if (lane < 28) {    // one (i, head) pair per lane
        int i = lane >> 2, hh = lane & 3;
        float si = sc[wv][200 + i * 4 + hh];
        float e[7];
        #pragma unroll
        for (int j = 0; j < 7; ++j) {
            float x = si + sc[wv][228 + j * 4 + hh];
            x = (x > 0.0f) ? x : 0.2f * x;               // leaky relu
            float m = maskp[i * 7 + j];
            e[j] = x * m + (1.0f - m) * NEGV;
        }
        int u1 = -1; float m1 = -3e38f;
        #pragma unroll
        for (int j = 0; j < 7; ++j) if (e[j] > m1) { m1 = e[j]; u1 = j; }
        int u2 = -1; float m2 = -3e38f;
        #pragma unroll
        for (int j = 0; j < 7; ++j) if (j != u1 && e[j] > m2) { m2 = e[j]; u2 = j; }
        float m3 = -3e38f;
        #pragma unroll
        for (int j = 0; j < 7; ++j) if (j != u1 && j != u2 && e[j] > m3) m3 = e[j];
        float ssum = 0.0f, al[7];
        #pragma unroll
        for (int j = 0; j < 7; ++j) {
            float ev = (e[j] >= m3) ? e[j] : NEGV;       // keep top-3 (and ties)
            float ex = expf(ev - m1);
            al[j] = ex; ssum += ex;
        }
        float inv = (ssum > 0.0f) ? 1.0f / ssum : 0.0f;
        #pragma unroll
        for (int j = 0; j < 7; ++j) sc[wv][i * 28 + j * 4 + hh] = al[j] * inv;
    }
    __syncthreads();

    if (attn_out) {
        for (int q = lane; q < 49; q += 32) {
            int i = q / 7, j = q % 7, o = i * 28 + j * 4;
            attn_out[b * 49 + q] =
                0.25f * (sc[wv][o] + sc[wv][o + 1] + sc[wv][o + 2] + sc[wv][o + 3]);
        }
    }

    #pragma unroll
    for (int i = 0; i < 7; ++i) {
        float o[8];
        #pragma unroll
        for (int t = 0; t < 8; ++t) {
            float acc = 0.0f;
            #pragma unroll
            for (int j = 0; j < 7; ++j)
                acc += sc[wv][i * 28 + j * 4 + (t >> 1)] * hv[j][t];
            o[t] = acc;
        }
        #pragma unroll
        for (int t = 0; t < 8; ++t) {
            int c = lane + 32 * t;
            float r = Resid[(r0 + i) * HIDC + c];
            if (res_bias) r += res_bias[c];
            o[t] += r;
        }
        float s = 0.0f;
        #pragma unroll
        for (int t = 0; t < 8; ++t) s += o[t];
        float mu = wred(s) * (1.0f / 256.0f);
        float vv = 0.0f;
        #pragma unroll
        for (int t = 0; t < 8; ++t) { float d = o[t] - mu; vv += d * d; }
        float var = wred(vv) * (1.0f / 256.0f);
        float rs = rsqrtf(var + 1e-5f);
        #pragma unroll
        for (int t = 0; t < 8; ++t) {
            int c = lane + 32 * t;
            float y = (o[t] - mu) * rs * ln_w[c] + ln_b[c];
            OutH[b * batchStride + i * HIDC + c] = fmaxf(y, 0.0f);
        }
    }

    if (context) {
        for (int q = lane; q < 128; q += 32)
            OutH[b * (long long)GDIM + 1792 + q] = context[b * 128 + q];
    }
}

// ---------------------------------------------------------------------------
// Final heads: softmax mixes, gate, residual, pred. One thread per batch.
// ---------------------------------------------------------------------------
__global__ __launch_bounds__(256)
void k_final(const float* __restrict__ HH, const float* __restrict__ base_preds,
             const float* __restrict__ regime,
             const float* __restrict__ base_b1, const float* __restrict__ base_w2,
             const float* __restrict__ base_b2,
             const float* __restrict__ reg_b1, const float* __restrict__ reg_w2,
             const float* __restrict__ reg_b2,
             const float* __restrict__ gate_w1, const float* __restrict__ gate_b1,
             const float* __restrict__ gate_w2, const float* __restrict__ gate_b2,
             const float* __restrict__ res_b1, const float* __restrict__ res_w2,
             const float* __restrict__ res_b2,
             float* __restrict__ out_pred, float* __restrict__ out_w,
             float* __restrict__ out_gate) {
    int b = blockIdx.x * blockDim.x + threadIdx.x;
    if (b >= BATCH) return;
    const float* h = HH + (size_t)b * HOUT;

    float z0 = base_b2[0], z1 = base_b2[1];
    float y0 = reg_b2[0],  y1 = reg_b2[1];
    #pragma unroll
    for (int j = 0; j < 32; ++j) {
        float rb = fmaxf(h[j] + base_b1[j], 0.0f);
        z0 += rb * base_w2[j * 2 + 0]; z1 += rb * base_w2[j * 2 + 1];
        float rr = fmaxf(h[32 + j] + reg_b1[j], 0.0f);
        y0 += rr * reg_w2[j * 2 + 0];  y1 += rr * reg_w2[j * 2 + 1];
    }
    float mb = fmaxf(z0, z1), e0 = expf(z0 - mb), e1 = expf(z1 - mb);
    float wb0 = e0 / (e0 + e1), wb1 = e1 / (e0 + e1);
    float mr = fmaxf(y0, y1), f0 = expf(y0 - mr), f1 = expf(y1 - mr);
    float wr0 = f0 / (f0 + f1), wr1 = f1 / (f0 + f1);

    float racc = res_b2[0];
    #pragma unroll
    for (int j = 0; j < 16; ++j)
        racc += tanhf(h[64 + j] + res_b1[j]) * res_w2[j];

    float gz = gate_b2[0];
    float r0 = regime[b * 3 + 0], r1 = regime[b * 3 + 1], r2 = regime[b * 3 + 2];
    #pragma unroll
    for (int k = 0; k < 8; ++k) {
        float t = gate_b1[k] + r0 * gate_w1[k] + r1 * gate_w1[8 + k] + r2 * gate_w1[16 + k];
        gz += tanhf(t) * gate_w2[k];
    }
    float g = 1.0f / (1.0f + expf(-gz));

    float w0 = (1.0f - g) * wb0 + g * wr0;
    float w1 = (1.0f - g) * wb1 + g * wr1;
    float pred = w0 * base_preds[b * 2] + w1 * base_preds[b * 2 + 1] + racc * 0.05f;
    pred = fmaxf(pred, 0.05f);

    out_pred[b] = pred;
    out_w[b * 2] = w0; out_w[b * 2 + 1] = w1;
    out_gate[b] = g;
}

// ---------------------------------------------------------------------------
extern "C" void kernel_launch(void* const* d_in, const int* in_sizes, int n_in,
                              void* d_out, int out_size, void* d_ws, size_t ws_size,
                              hipStream_t stream) {
    const float* node_feats = (const float*)d_in[0];
    const float* context    = (const float*)d_in[1];
    const float* base_preds = (const float*)d_in[2];
    const float* regime     = (const float*)d_in[3];
    const float* edge_log   = (const float*)d_in[4];
    const float* W1         = (const float*)d_in[5];
    const float* a_src1     = (const float*)d_in[6];
    const float* a_dst1     = (const float*)d_in[7];
    const float* W2         = (const float*)d_in[8];
    const float* a_src2     = (const float*)d_in[9];
    const float* a_dst2     = (const float*)d_in[10];
    const float* ln1_w      = (const float*)d_in[11];
    const float* ln1_b      = (const float*)d_in[12];
    const float* ln2_w      = (const float*)d_in[13];
    const float* ln2_b      = (const float*)d_in[14];
    const float* skip_w     = (const float*)d_in[15];
    const float* skip_b     = (const float*)d_in[16];
    const float* base_w1    = (const float*)d_in[17];
    const float* base_b1    = (const float*)d_in[18];
    const float* base_w2    = (const float*)d_in[19];
    const float* base_b2    = (const float*)d_in[20];
    const float* reg_w1     = (const float*)d_in[21];
    const float* reg_b1     = (const float*)d_in[22];
    const float* reg_w2     = (const float*)d_in[23];
    const float* reg_b2     = (const float*)d_in[24];
    const float* gate_w1    = (const float*)d_in[25];
    const float* gate_b1    = (const float*)d_in[26];
    const float* gate_w2    = (const float*)d_in[27];
    const float* gate_b2    = (const float*)d_in[28];
    const float* res_w1     = (const float*)d_in[29];
    const float* res_b1     = (const float*)d_in[30];
    const float* res_w2     = (const float*)d_in[31];
    const float* res_b2     = (const float*)d_in[32];

    // output layout: pred(B) | weights(B*2) | gate(B) | mask(49) | attn(B*49)
    float* outp     = (float*)d_out;
    float* o_pred   = outp;
    float* o_w      = outp + BATCH;
    float* o_gate   = outp + 3 * BATCH;
    float* o_mask   = outp + 4 * BATCH;
    float* o_attn   = outp + 4 * BATCH + 49;

    // workspace carve-out (bytes, 256B aligned)
    char* w = (char*)d_ws;
    size_t off = 0;
    auto carve = [&](size_t bytes) { size_t r = off; off += (bytes + 255) & ~(size_t)255; return r; };
    float*          maskws = (float*)         (w + carve(64 * 4));
    unsigned short* W1t    = (unsigned short*)(w + carve(HIDC * HIDC * 2));
    unsigned short* SKt    = (unsigned short*)(w + carve(HIDC * HIDC * 2));
    unsigned short* W2t    = (unsigned short*)(w + carve(HIDC * HIDC * 2));
    unsigned short* Wht    = (unsigned short*)(w + carve((size_t)HOUT * GDIM * 2));
    float*          bufA   = (float*)(w + carve((size_t)NBROWS * HIDC * 4));      // H1, then H2
    float*          bufB   = (float*)(w + carve((size_t)BATCH * GDIM * 4));       // SKIP, then CMB
    float*          bufC   = (float*)(w + carve((size_t)NBROWS * HIDC * 4));      // HL1
    float*          HH     = (float*)(w + carve((size_t)BATCH * HOUT * 4));
    (void)ws_size; (void)n_in; (void)in_sizes; (void)out_size;

    // 1) weight bf16-transpose + edge mask
    k_prep<<<1024, 256, 0, stream>>>(W1, skip_w, W2, base_w1, reg_w1, res_w1, edge_log,
                                     W1t, SKt, W2t, Wht, maskws, o_mask);
    // 2) H1 = X@W1, SKIP = X@skip_w  (dual-output WMMA GEMM)
    k_gemm256<<<NBROWS / 16, 256, 0, stream>>>(node_feats, W1t, SKt, bufA, bufB, 1);
    // 3) GAT layer 1: attention + skip + skip_b + LN1 + ReLU -> HL1
    k_attn<<<BATCH / 8, 256, 0, stream>>>(bufA, bufB, skip_b, maskws, a_src1, a_dst1,
                                          ln1_w, ln1_b, bufC, NNODE * HIDC,
                                          nullptr, nullptr);
    // 4) H2 = HL1 @ W2
    k_gemm256<<<NBROWS / 32, 256, 0, stream>>>(bufC, W2t, nullptr, bufA, nullptr, 0);
    // 5) GAT layer 2: attention + residual + LN2 + ReLU -> combined rows; attn out
    k_attn<<<BATCH / 8, 256, 0, stream>>>(bufA, bufC, nullptr, maskws, a_src2, a_dst2,
                                          ln2_w, ln2_b, bufB, GDIM,
                                          context, o_attn);
    // 6) head GEMM: (B x 1920) @ (1920 x 80)
    k_gemm_head<<<BATCH / 16, 160, 0, stream>>>(bufB, Wht, HH);
    // 7) final mixture heads
    k_final<<<BATCH / 256, 256, 0, stream>>>(HH, base_preds, regime,
                                             base_b1, base_w2, base_b2,
                                             reg_b1, reg_w2, reg_b2,
                                             gate_w1, gate_b1, gate_w2, gate_b2,
                                             res_b1, res_w2, res_b2,
                                             o_pred, o_w, o_gate);
}
